// ResidualRefiner_5317169512781
// MI455X (gfx1250) — matrix operand order repaired and checked by reading, not compiled, not run on hardware
//
#include <hip/hip_runtime.h>
#include <hip/hip_bf16.h>
#include <math.h>

// ---------------------------------------------------------------------------
// ResidualRefiner forward for MI455X (gfx1250, wave32, WMMA bf16).
// GEMMs use v_wmma_f32_16x16x32_bf16 (fp32 accumulate), BN folded into
// epilogue scale/bias.  Attention rewritten associatively:
//   val = (q^T k) v^T  ==  (v k^T) q     (no softmax -> exact)
// coord-weight einsum collapses to a row-sum vector dot.
//
// Activation tensors consumed as the B operand are stored K-pair packed
// (uint = channels {2c,2c+1}) so both global->LDS staging and LDS->fragment
// loads are contiguous 32/128-bit ops matching the CDNA5 WMMA VGPR layout.
// ---------------------------------------------------------------------------

typedef __attribute__((ext_vector_type(16))) __bf16 v16bf;
typedef __attribute__((ext_vector_type(2)))  __bf16 v2bf;
typedef __attribute__((ext_vector_type(8)))  float  v8f;
typedef __attribute__((ext_vector_type(8)))  unsigned int v8u;

#define NPTS  4096
#define NBATCH 4
#define CCH   64
#define KNN   16
#define NKCOL (NPTS * KNN)   // 65536
#define EPSBN 1e-5f

__device__ __forceinline__ unsigned int pk2(float a, float b) {
  v2bf t; t[0] = (__bf16)a; t[1] = (__bf16)b;
  return __builtin_bit_cast(unsigned int, t);
}
// read channel c, column n from a K-pair packed (C/2 x Ncols uint) tensor
__device__ __forceinline__ float ldpk(const __bf16* buf, int c, int n, int ncols) {
  unsigned int u = ((const unsigned int*)buf)[(size_t)(c >> 1) * ncols + n];
  unsigned short h = (c & 1) ? (unsigned short)(u >> 16) : (unsigned short)(u & 0xffffu);
  return (float)__builtin_bit_cast(__bf16, h);
}

// ---------------------------------------------------------------------------
// Weight prep: fp32 (Cout x Cin) -> bf16 (Cout x CinPad), zero padded K.
// ---------------------------------------------------------------------------
__global__ void k_prep_w(const float* __restrict__ W, __bf16* __restrict__ o,
                         int Cout, int Cin, int CinPad) {
  int i = blockIdx.x * 256 + threadIdx.x;
  if (i >= Cout * CinPad) return;
  int m = i / CinPad, kk = i - m * CinPad;
  o[i] = (__bf16)(kk < Cin ? W[m * Cin + kk] : 0.0f);
}

// BN fold: scale = g*rsqrt(v+eps); bias = (conv_b - mean)*scale + beta.
__global__ void k_prep_sb(const float* g, const float* be, const float* mu,
                          const float* va, const float* cb,
                          float* sc, float* bi, int C) {
  int m = blockIdx.x * 64 + threadIdx.x;
  if (m >= C) return;
  float s = 1.0f, b = cb ? cb[m] : 0.0f;
  if (g) { s = g[m] * rsqrtf(va[m] + EPSBN); b = (b - mu[m]) * s + be[m]; }
  sc[m] = s; bi[m] = b;
}

// wrow[c] = sum_k coord_W2[k][c]  (16x64), bsum = sum coord_b2
__global__ void k_prep_wrow(const float* __restrict__ W2,
                            const float* __restrict__ b2,
                            float* __restrict__ wrow, float* __restrict__ bsum) {
  int c = threadIdx.x;           // 64 threads
  float s = 0.0f;
  for (int k = 0; k < 16; ++k) s += W2[k * 64 + c];
  wrow[c] = s;
  if (c == 0) {
    float bs = 0.0f;
    for (int k = 0; k < 16; ++k) bs += b2[k];
    *bsum = bs;
  }
}

// ---------------------------------------------------------------------------
// KNN: all 4096 coords staged in LDS (48 KB of 320 KB); 17-entry insertion
// sort per lane, drop self, emit 16 neighbor indices.
// ---------------------------------------------------------------------------
__global__ __launch_bounds__(256) void k_knn(const float* __restrict__ xyz,
                                             int* __restrict__ idx, int b) {
  __shared__ float sx[NPTS], sy[NPTS], sz[NPTS];
  const float* p = xyz + (size_t)b * 3 * NPTS;
  for (int i = threadIdx.x; i < NPTS; i += 256) {
    sx[i] = p[i]; sy[i] = p[NPTS + i]; sz[i] = p[2 * NPTS + i];
  }
  __syncthreads();
  int n = blockIdx.x * 256 + threadIdx.x;
  float qx = sx[n], qy = sy[n], qz = sz[n];
  float bd[KNN + 1]; int bi[KNN + 1];
#pragma unroll
  for (int i = 0; i <= KNN; ++i) { bd[i] = 3.4e38f; bi[i] = 0; }
  for (int m = 0; m < NPTS; ++m) {
    float dx = sx[m] - qx, dy = sy[m] - qy, dz = sz[m] - qz;
    float d = dx * dx + dy * dy + dz * dz;
    if (d < bd[KNN]) {
      int pnt = KNN;
      while (pnt > 0 && bd[pnt - 1] > d) { bd[pnt] = bd[pnt - 1]; bi[pnt] = bi[pnt - 1]; --pnt; }
      bd[pnt] = d; bi[pnt] = m;
    }
  }
  int* o = idx + (size_t)n * KNN;
  for (int j = 1; j <= KNN; ++j) o[j - 1] = bi[j];   // drop self (nearest)
}

// ---------------------------------------------------------------------------
// Build grouped activations, K-pair packed:
//   cat channels 0..63 = gathered features, 64..66 = diff, 67..95 = 0
//   diffp channels 0..2 = diff, 3..31 = 0
// ---------------------------------------------------------------------------
__global__ __launch_bounds__(256) void k_build_cat(
    const float* __restrict__ x, const float* __restrict__ xyz,
    const int* __restrict__ idx, __bf16* __restrict__ cat,
    __bf16* __restrict__ diffp, int b) {
  int col = blockIdx.x * 256 + threadIdx.x;          // 0..NKCOL-1
  int n = col >> 4;
  int nb = idx[col];
  const float* p = xyz + (size_t)b * 3 * NPTS;
  float dx = p[nb] - p[n];
  float dy = p[NPTS + nb] - p[NPTS + n];
  float dz = p[2 * NPTS + nb] - p[2 * NPTS + n];
  const float* fx = x + (size_t)b * CCH * NPTS;
  unsigned int* catp = (unsigned int*)cat;
  for (int cp = 0; cp < 32; ++cp)
    catp[(size_t)cp * NKCOL + col] =
        pk2(fx[(size_t)(2 * cp) * NPTS + nb], fx[(size_t)(2 * cp + 1) * NPTS + nb]);
  catp[(size_t)32 * NKCOL + col] = pk2(dx, dy);
  catp[(size_t)33 * NKCOL + col] = pk2(dz, 0.0f);
  for (int cp = 34; cp < 48; ++cp) catp[(size_t)cp * NKCOL + col] = 0u;
  unsigned int* dfp = (unsigned int*)diffp;
  dfp[(size_t)0 * NKCOL + col] = pk2(dx, dy);
  dfp[(size_t)1 * NKCOL + col] = pk2(dz, 0.0f);
  for (int cp = 2; cp < 16; ++cp) dfp[(size_t)cp * NKCOL + col] = 0u;
}

// cat_g (96 x NPTS, packed): ch 0..2 = xyz, 3..66 = x, 67..95 = 0
__global__ __launch_bounds__(256) void k_build_catg(
    const float* __restrict__ x, const float* __restrict__ xyz,
    __bf16* __restrict__ catg, int b) {
  int n = blockIdx.x * 256 + threadIdx.x;
  const float* p = xyz + (size_t)b * 3 * NPTS;
  const float* fx = x + (size_t)b * CCH * NPTS;
  unsigned int* cp_ = (unsigned int*)catg;
  auto val = [&](int c) -> float {
    return c < 3 ? p[(size_t)c * NPTS + n]
                 : (c < 67 ? fx[(size_t)(c - 3) * NPTS + n] : 0.0f);
  };
  for (int cp = 0; cp < 48; ++cp)
    cp_[(size_t)cp * NPTS + n] = pk2(val(2 * cp), val(2 * cp + 1));
}

// ---------------------------------------------------------------------------
// WMMA GEMM, tile 32(M) x 128(N), 128 threads (4 waves), 4 WMMAs per K-step:
//   out(Cout x Ncols) = act( scale .* (W(Cout x K) @ X) + bias )
// W: plain bf16 row-major (K contiguous).
// X: K-pair packed (K/2 x Ncols uint) normally; plain (Ncols x K) if btrans.
// Fragment feeds are single ds_load_b32 per VGPR (pair layout):
//   A VGPR v (half h, row lane%16): pair = v + 4h + (v>=4 ? 4:0)
//   B VGPR v (half h, col lane%16): pair = v + 4h + (v>=4 ? 4:0)
//   C/D VGPR r: M = r + 8h, N = lane%16.
// Staging is register-buffered and fully 128-bit: A = 1 b128 load + 1 b128
// LDS store per thread (row stride padded to 20 uints = 80 B: 16B-aligned and
// bank-conflict-free), B = 4 b128 loads + stores.
// ---------------------------------------------------------------------------
__global__ __launch_bounds__(128) void k_gemm_bf16(
    const __bf16* __restrict__ W, const __bf16* __restrict__ X,
    const float* __restrict__ scale, const float* __restrict__ bias,
    __bf16* __restrict__ outb, float* __restrict__ outf,
    int Cout, int K, int Ncols, int act, int btrans, int addf, int pack) {
  __shared__ unsigned int Asm32[32][20];    // [m][kpair], padded stride (80 B)
  __shared__ unsigned int Bsm32[16][128];   // [kpair][col]
  const int tid  = threadIdx.x;
  const int lane = tid & 31;
  const int wave = tid >> 5;
  const int half = lane >> 4;
  const int l15  = lane & 15;
  const int mBase = blockIdx.y * 32;
  const int nBase = blockIdx.x * 128;
  const int ra = tid >> 2, qa = (tid & 3) << 2;      // A granule: row, pair-quad
  v8f acc00 = {}, acc01 = {}, acc10 = {}, acc11 = {};
  for (int k0 = 0; k0 < K; k0 += 32) {
    // ---- stage A tile: 32 rows x 16 K-pairs as one uint4 per thread ----
    uint4 ta = *(const uint4*)&W[(size_t)(mBase + ra) * K + k0 + 2 * qa];
    // ---- stage B tile: 16 K-pairs x 128 cols (2048 uints) ----
    if (btrans) {
      unsigned int tb[16];
#pragma unroll
      for (int j = 0; j < 16; ++j) {
        int i = tid + j * 128, r = i >> 7, c = i & 127;
        tb[j] = *(const unsigned int*)&X[(size_t)(nBase + c) * K + k0 + 2 * r];
      }
      *(uint4*)&Asm32[ra][qa] = ta;
#pragma unroll
      for (int j = 0; j < 16; ++j) {
        int i = tid + j * 128, r = i >> 7, c = i & 127;
        Bsm32[r][c] = tb[j];
      }
    } else {
      const uint4* Xq = (const uint4*)X;      // packed: (K/2) x (Ncols/4) uint4
      const int nq = Ncols >> 2;
      const int kp0 = k0 >> 1;
      uint4 tb[4];
#pragma unroll
      for (int j = 0; j < 4; ++j) {
        int i = tid + j * 128, r = i >> 5, c4 = i & 31;
        tb[j] = Xq[(size_t)(kp0 + r) * nq + (nBase >> 2) + c4];
      }
      *(uint4*)&Asm32[ra][qa] = ta;
#pragma unroll
      for (int j = 0; j < 4; ++j) {
        int i = tid + j * 128, r = i >> 5, c4 = i & 31;
        *(uint4*)&Bsm32[r][c4 << 2] = tb[j];
      }
      if (k0 + 32 < K)
        __builtin_prefetch(&Xq[(size_t)(kp0 + 16) * nq + (nBase >> 2) + (tid & 31)], 0, 0);
    }
    __syncthreads();
    v8u au0, au1, bu0, bu1;
#pragma unroll
    for (int v = 0; v < 8; ++v) {
      int pa = v + half * 4 + (v >= 4 ? 4 : 0);
      au0[v] = Asm32[l15][pa];
      au1[v] = Asm32[16 + l15][pa];
      int pb = v + half * 4 + (v >= 4 ? 4 : 0);
      bu0[v] = Bsm32[pb][wave * 16 + l15];
      bu1[v] = Bsm32[pb][64 + wave * 16 + l15];
    }
    v16bf a0 = __builtin_bit_cast(v16bf, au0);
    v16bf a1 = __builtin_bit_cast(v16bf, au1);
    v16bf b0 = __builtin_bit_cast(v16bf, bu0);
    v16bf b1 = __builtin_bit_cast(v16bf, bu1);
    acc00 = __builtin_amdgcn_wmma_f32_16x16x32_bf16(false, a0, false, b0, (short)0, acc00, false, false);
    acc01 = __builtin_amdgcn_wmma_f32_16x16x32_bf16(false, a0, false, b1, (short)0, acc01, false, false);
    acc10 = __builtin_amdgcn_wmma_f32_16x16x32_bf16(false, a1, false, b0, (short)0, acc10, false, false);
    acc11 = __builtin_amdgcn_wmma_f32_16x16x32_bf16(false, a1, false, b1, (short)0, acc11, false, false);
    __syncthreads();
  }
  // ---- epilogue ----
  const int n0 = nBase + wave * 16 + l15;
#pragma unroll
  for (int ms = 0; ms < 2; ++ms) {
    const v8f& c0 = ms ? acc10 : acc00;
    const v8f& c1 = ms ? acc11 : acc01;
    float f0[8], f1[8];
#pragma unroll
    for (int r = 0; r < 8; ++r) {
      int m = mBase + ms * 16 + half * 8 + r;
      float sc = scale ? scale[m] : 1.0f;
      float bi = bias ? bias[m] : 0.0f;
      float a = c0[r] * sc + bi, b = c1[r] * sc + bi;
      if (act == 1) { a = fmaxf(a, 0.0f); b = fmaxf(b, 0.0f); }
      f0[r] = a; f1[r] = b;
      if (outf) {
        size_t o = (size_t)m * Ncols + n0;
        if (addf) { outf[o] += a; outf[o + 64] += b; }
        else      { outf[o]  = a; outf[o + 64]  = b; }
      }
    }
    if (outb) {
      if (pack) {
        unsigned int* ob = (unsigned int*)outb;
#pragma unroll
        for (int r = 0; r < 8; r += 2) {
          int mp = (mBase + ms * 16 + half * 8 + r) >> 1;
          ob[(size_t)mp * Ncols + n0]      = pk2(f0[r], f0[r + 1]);
          ob[(size_t)mp * Ncols + n0 + 64] = pk2(f1[r], f1[r + 1]);
        }
      } else {
#pragma unroll
        for (int r = 0; r < 8; ++r) {
          int m = mBase + ms * 16 + half * 8 + r;
          outb[(size_t)m * Ncols + n0]      = (__bf16)f0[r];
          outb[(size_t)m * Ncols + n0 + 64] = (__bf16)f1[r];
        }
      }
    }
  }
}

// ---------------------------------------------------------------------------
// wsumcol[col] = bsum + sum_c wrow[c] * c1[c][col]   (c1 packed)
// ---------------------------------------------------------------------------
__global__ __launch_bounds__(256) void k_wsumcol(
    const __bf16* __restrict__ c1, const float* __restrict__ wrow,
    const float* __restrict__ bsum, float* __restrict__ wsc) {
  int col = blockIdx.x * 256 + threadIdx.x;
  const unsigned int* p = (const unsigned int*)c1;
  float s = *bsum;
  for (int cp = 0; cp < 32; ++cp) {
    v2bf t = __builtin_bit_cast(v2bf, p[(size_t)cp * NKCOL + col]);
    s += wrow[2 * cp] * (float)t[0] + wrow[2 * cp + 1] * (float)t[1];
  }
  wsc[col] = s;
}

// local[c][n] = sum_j wsc[n*16+j]*feat[c][n*16+j] + max_j mp2[c][n*16+j]
__global__ __launch_bounds__(256) void k_local(
    const float* __restrict__ wsc, const __bf16* __restrict__ feat,
    const __bf16* __restrict__ mp2, float* __restrict__ rF) {
  int n = blockIdx.x * 256 + threadIdx.x;
  int c = blockIdx.y;
  float acc = 0.0f, mx = -3.4e38f;
#pragma unroll
  for (int j = 0; j < KNN; ++j) {
    int col = n * KNN + j;
    acc += wsc[col] * ldpk(feat, c, col, NKCOL);
    mx = fmaxf(mx, ldpk(mp2, c, col, NKCOL));
  }
  rF[(size_t)c * NPTS + n] = acc + mx;
}

// rF (64 x NPTS f32) -> packed bf16
__global__ __launch_bounds__(256) void k_pack_r(
    const float* __restrict__ rF, __bf16* __restrict__ rB) {
  int i = blockIdx.x * 256 + threadIdx.x;     // over 32*NPTS uints
  int cp = i / NPTS, n = i - cp * NPTS;
  ((unsigned int*)rB)[(size_t)cp * NPTS + n] =
      pk2(rF[(size_t)(2 * cp) * NPTS + n], rF[(size_t)(2 * cp + 1) * NPTS + n]);
}

// Final 64->3 layer + sigmoid - 0.5 (too narrow for WMMA); o2 packed.
__global__ __launch_bounds__(256) void k_final3(
    const __bf16* __restrict__ o2, const float* __restrict__ W3,
    const float* __restrict__ b3, float* __restrict__ out, int b) {
  int n = blockIdx.x * 256 + threadIdx.x;
  const unsigned int* p = (const unsigned int*)o2;
  float col[CCH];
  for (int cp = 0; cp < 32; ++cp) {
    v2bf t = __builtin_bit_cast(v2bf, p[(size_t)cp * NPTS + n]);
    col[2 * cp] = (float)t[0]; col[2 * cp + 1] = (float)t[1];
  }
#pragma unroll
  for (int d = 0; d < 3; ++d) {
    float a = b3[d];
    for (int c = 0; c < CCH; ++c) a += W3[d * CCH + c] * col[c];
    out[(size_t)b * 3 * NPTS + (size_t)d * NPTS + n] =
        1.0f / (1.0f + expf(-a)) - 0.5f;
  }
}

// ---------------------------------------------------------------------------
// Host orchestration
// ---------------------------------------------------------------------------
extern "C" void kernel_launch(void* const* d_in, const int* in_sizes, int n_in,
                              void* d_out, int out_size, void* d_ws, size_t ws_size,
                              hipStream_t stream) {
  (void)in_sizes; (void)n_in; (void)out_size; (void)ws_size;
  const float* x    = (const float*)d_in[0];
  const float* xyz  = (const float*)d_in[1];
  const float* feat_W1 = (const float*)d_in[2];
  const float* f1g = (const float*)d_in[3]; const float* f1b = (const float*)d_in[4];
  const float* f1m = (const float*)d_in[5]; const float* f1v = (const float*)d_in[6];
  const float* feat_W2 = (const float*)d_in[7];
  const float* feat_b2 = (const float*)d_in[8];
  const float* f2g = (const float*)d_in[9];  const float* f2b = (const float*)d_in[10];
  const float* f2m = (const float*)d_in[11]; const float* f2v = (const float*)d_in[12];
  const float* coord_W1 = (const float*)d_in[13];
  const float* c1g = (const float*)d_in[14]; const float* c1b = (const float*)d_in[15];
  const float* c1m = (const float*)d_in[16]; const float* c1v = (const float*)d_in[17];
  const float* coord_W2 = (const float*)d_in[18];
  const float* coord_b2 = (const float*)d_in[19];
  const float* mp_W1 = (const float*)d_in[20];
  const float* m1g = (const float*)d_in[21]; const float* m1b = (const float*)d_in[22];
  const float* m1m = (const float*)d_in[23]; const float* m1v = (const float*)d_in[24];
  const float* mp_W2 = (const float*)d_in[25];
  const float* m2g = (const float*)d_in[26]; const float* m2b = (const float*)d_in[27];
  const float* m2m = (const float*)d_in[28]; const float* m2v = (const float*)d_in[29];
  const float* q_W = (const float*)d_in[30];
  const float* qg = (const float*)d_in[31]; const float* qb = (const float*)d_in[32];
  const float* qm = (const float*)d_in[33]; const float* qv = (const float*)d_in[34];
  const float* k_W = (const float*)d_in[35];
  const float* kg = (const float*)d_in[36]; const float* kb = (const float*)d_in[37];
  const float* km = (const float*)d_in[38]; const float* kv = (const float*)d_in[39];
  const float* v_W = (const float*)d_in[40];
  const float* vg = (const float*)d_in[41]; const float* vb = (const float*)d_in[42];
  const float* vm = (const float*)d_in[43]; const float* vv = (const float*)d_in[44];
  const float* fus_W = (const float*)d_in[45];
  const float* fg = (const float*)d_in[46]; const float* fb = (const float*)d_in[47];
  const float* fm = (const float*)d_in[48]; const float* fv = (const float*)d_in[49];
  const float* off_W1 = (const float*)d_in[50];
  const float* off_b1 = (const float*)d_in[51];
  const float* off_W2 = (const float*)d_in[52];
  const float* off_b2 = (const float*)d_in[53];
  const float* off_W3 = (const float*)d_in[54];
  const float* off_b3 = (const float*)d_in[55];
  float* out = (float*)d_out;

  char* ws = (char*)d_ws;
  size_t cur = 0;
  auto alloc = [&](size_t bytes) -> void* {
    void* p = ws + cur;
    cur += (bytes + 255) & ~(size_t)255;
    return p;
  };

  // prepped weights (bf16, K padded, plain row-major: consumed as A operand)
  __bf16* wFeat1 = (__bf16*)alloc(128 * 96 * 2);
  __bf16* wFeat2 = (__bf16*)alloc(64 * 128 * 2);
  __bf16* wCoord1 = (__bf16*)alloc(64 * 32 * 2);
  __bf16* wMp1 = (__bf16*)alloc(128 * 64 * 2);
  __bf16* wMp2 = (__bf16*)alloc(64 * 128 * 2);
  __bf16* wQ = (__bf16*)alloc(128 * 96 * 2);
  __bf16* wK = (__bf16*)alloc(128 * 96 * 2);
  __bf16* wV = (__bf16*)alloc(128 * 96 * 2);
  __bf16* wFus = (__bf16*)alloc(64 * 128 * 2);
  __bf16* wOff1 = (__bf16*)alloc(256 * 64 * 2);
  __bf16* wOff2 = (__bf16*)alloc(64 * 256 * 2);
  // folded scale/bias
  float* sF1 = (float*)alloc(128 * 4); float* bF1 = (float*)alloc(128 * 4);
  float* sF2 = (float*)alloc(64 * 4);  float* bF2 = (float*)alloc(64 * 4);
  float* sC1 = (float*)alloc(64 * 4);  float* bC1 = (float*)alloc(64 * 4);
  float* sM1 = (float*)alloc(128 * 4); float* bM1 = (float*)alloc(128 * 4);
  float* sM2 = (float*)alloc(64 * 4);  float* bM2 = (float*)alloc(64 * 4);
  float* sQ = (float*)alloc(128 * 4);  float* bQ = (float*)alloc(128 * 4);
  float* sK = (float*)alloc(128 * 4);  float* bK = (float*)alloc(128 * 4);
  float* sV = (float*)alloc(128 * 4);  float* bV = (float*)alloc(128 * 4);
  float* sFu = (float*)alloc(64 * 4);  float* bFu = (float*)alloc(64 * 4);
  float* wrow = (float*)alloc(64 * 4); float* bsum = (float*)alloc(4);
  // per-batch activations (reused across batches); packed unless noted
  int* idxBuf = (int*)alloc((size_t)NPTS * KNN * 4);
  __bf16* catBuf  = (__bf16*)alloc((size_t)96 * NKCOL * 2);
  __bf16* diffBuf = (__bf16*)alloc((size_t)32 * NKCOL * 2);
  __bf16* hBuf    = (__bf16*)alloc((size_t)128 * NKCOL * 2);   // also mp1
  __bf16* featBuf = (__bf16*)alloc((size_t)64 * NKCOL * 2);
  __bf16* mp2Buf  = (__bf16*)alloc((size_t)64 * NKCOL * 2);
  __bf16* c1Buf   = (__bf16*)alloc((size_t)64 * NKCOL * 2);
  float*  wscBuf  = (float*)alloc((size_t)NKCOL * 4);
  __bf16* catG = (__bf16*)alloc((size_t)96 * NPTS * 2);
  __bf16* qBuf = (__bf16*)alloc((size_t)128 * NPTS * 2);
  __bf16* kBuf = (__bf16*)alloc((size_t)128 * NPTS * 2);       // plain
  __bf16* vBuf = (__bf16*)alloc((size_t)128 * NPTS * 2);       // plain
  __bf16* GBuf = (__bf16*)alloc((size_t)128 * 128 * 2);        // plain
  __bf16* valBuf = (__bf16*)alloc((size_t)128 * NPTS * 2);
  float*  rF = (float*)alloc((size_t)64 * NPTS * 4);
  __bf16* rB = (__bf16*)alloc((size_t)64 * NPTS * 2);
  __bf16* o1Buf = (__bf16*)alloc((size_t)256 * NPTS * 2);
  __bf16* o2Buf = (__bf16*)alloc((size_t)64 * NPTS * 2);

  auto prepw = [&](const float* W, __bf16* o, int Cout, int Cin, int CinPad) {
    int tot = Cout * CinPad;
    k_prep_w<<<(tot + 255) / 256, 256, 0, stream>>>(W, o, Cout, Cin, CinPad);
  };
  prepw(feat_W1, wFeat1, 128, 67, 96);
  prepw(feat_W2, wFeat2, 64, 128, 128);
  prepw(coord_W1, wCoord1, 64, 3, 32);
  prepw(mp_W1, wMp1, 128, 64, 64);
  prepw(mp_W2, wMp2, 64, 128, 128);
  prepw(q_W, wQ, 128, 67, 96);
  prepw(k_W, wK, 128, 67, 96);
  prepw(v_W, wV, 128, 67, 96);
  prepw(fus_W, wFus, 64, 128, 128);
  prepw(off_W1, wOff1, 256, 64, 64);
  prepw(off_W2, wOff2, 64, 256, 256);
  k_prep_sb<<<2, 64, 0, stream>>>(f1g, f1b, f1m, f1v, nullptr, sF1, bF1, 128);
  k_prep_sb<<<1, 64, 0, stream>>>(f2g, f2b, f2m, f2v, feat_b2, sF2, bF2, 64);
  k_prep_sb<<<1, 64, 0, stream>>>(c1g, c1b, c1m, c1v, nullptr, sC1, bC1, 64);
  k_prep_sb<<<2, 64, 0, stream>>>(m1g, m1b, m1m, m1v, nullptr, sM1, bM1, 128);
  k_prep_sb<<<1, 64, 0, stream>>>(m2g, m2b, m2m, m2v, nullptr, sM2, bM2, 64);
  k_prep_sb<<<2, 64, 0, stream>>>(qg, qb, qm, qv, nullptr, sQ, bQ, 128);
  k_prep_sb<<<2, 64, 0, stream>>>(kg, kb, km, kv, nullptr, sK, bK, 128);
  k_prep_sb<<<2, 64, 0, stream>>>(vg, vb, vm, vv, nullptr, sV, bV, 128);
  k_prep_sb<<<1, 64, 0, stream>>>(fg, fb, fm, fv, nullptr, sFu, bFu, 64);
  k_prep_wrow<<<1, 64, 0, stream>>>(coord_W2, coord_b2, wrow, bsum);

  auto gemm = [&](const __bf16* W, const __bf16* X, const float* sc,
                  const float* bi, __bf16* ob, float* of, int Cout, int K,
                  int Ncols, int act, int btrans, int addf, int pack) {
    dim3 grid(Ncols / 128, Cout / 32);
    k_gemm_bf16<<<grid, 128, 0, stream>>>(W, X, sc, bi, ob, of,
                                          Cout, K, Ncols, act, btrans, addf, pack);
  };

  for (int b = 0; b < NBATCH; ++b) {
    // ---- local branch (N*K = 65536 columns) ----
    k_knn<<<NPTS / 256, 256, 0, stream>>>(xyz, idxBuf, b);
    k_build_cat<<<NKCOL / 256, 256, 0, stream>>>(x, xyz, idxBuf, catBuf, diffBuf, b);
    gemm(wFeat1, catBuf, sF1, bF1, hBuf, nullptr, 128, 96, NKCOL, 1, 0, 0, 1);
    gemm(wFeat2, hBuf, sF2, bF2, featBuf, nullptr, 64, 128, NKCOL, 1, 0, 0, 1);
    gemm(wCoord1, diffBuf, sC1, bC1, c1Buf, nullptr, 64, 32, NKCOL, 1, 0, 0, 1);
    k_wsumcol<<<NKCOL / 256, 256, 0, stream>>>(c1Buf, wrow, bsum, wscBuf);
    gemm(wMp1, featBuf, sM1, bM1, hBuf, nullptr, 128, 64, NKCOL, 1, 0, 0, 1);
    gemm(wMp2, hBuf, sM2, bM2, mp2Buf, nullptr, 64, 128, NKCOL, 1, 0, 0, 1);
    k_local<<<dim3(NPTS / 256, 64), 256, 0, stream>>>(wscBuf, featBuf, mp2Buf, rF);

    // ---- global branch (associative linear attention) ----
    k_build_catg<<<NPTS / 256, 256, 0, stream>>>(x, xyz, catG, b);
    gemm(wQ, catG, sQ, bQ, qBuf, nullptr, 128, 96, NPTS, 1, 0, 0, 1);
    gemm(wK, catG, sK, bK, kBuf, nullptr, 128, 96, NPTS, 1, 0, 0, 0);  // plain
    gemm(wV, catG, sV, bV, vBuf, nullptr, 128, 96, NPTS, 1, 0, 0, 0);  // plain
    // G = v @ k^T (128 x 128, reduce over 4096 points): B-transposed mode
    gemm(vBuf, kBuf, nullptr, nullptr, GBuf, nullptr, 128, NPTS, 128, 0, 1, 0, 0);
    // val = G @ q (128 x 4096)
    gemm(GBuf, qBuf, nullptr, nullptr, valBuf, nullptr, 128, 128, NPTS, 0, 0, 0, 1);
    // glob = relu(bn(fus_W @ val)); r = local + glob (accumulate into rF)
    gemm(wFus, valBuf, sFu, bFu, nullptr, rF, 64, 128, NPTS, 1, 0, 1, 0);

    // ---- offset MLP ----
    k_pack_r<<<(32 * NPTS) / 256, 256, 0, stream>>>(rF, rB);
    gemm(wOff1, rB, nullptr, off_b1, o1Buf, nullptr, 256, 64, NPTS, 1, 0, 0, 1);
    gemm(wOff2, o1Buf, nullptr, off_b2, o2Buf, nullptr, 64, 256, NPTS, 1, 0, 0, 1);
    k_final3<<<NPTS / 256, 256, 0, stream>>>(o2Buf, off_W3, off_b3, out, b);
  }
}
